// Feature_extraction_layer_86526411145616
// MI455X (gfx1250) — matrix operand order repaired and checked by reading, not compiled, not run on hardware
//
#include <hip/hip_runtime.h>

typedef __attribute__((ext_vector_type(16))) _Float16 v16h;
typedef __attribute__((ext_vector_type(8)))  float    v8f;
typedef __attribute__((ext_vector_type(4)))  unsigned int u32x4;
typedef __attribute__((ext_vector_type(8)))  int      i32x8;
typedef __attribute__((ext_vector_type(4)))  int      i32x4;

#define BN 2
#define PN 8192
#define SN 8192
#define KN 16
#define SK (SN*KN)

#if defined(__has_builtin)
#if __has_builtin(__builtin_amdgcn_tensor_load_to_lds)
#define HAVE_TDM 1
#endif
#endif

__device__ __forceinline__ float lrelu(float x, float s){ return x >= 0.f ? x : s*x; }

// K index inside a 16-bit A-operand fragment (16x32), contiguous order matching B:
// element j of lane half h holds K = 16h + j
__device__ __forceinline__ int akidx(int j, int h){
  return 16*h + j;
}

// ---------------- 1. bitonic argsort of rand_num per batch ----------------
__global__ void k_sort(const float* __restrict__ rnd, int* __restrict__ perm) {
  __shared__ float key[PN];
  __shared__ int   kid[PN];
  const int b = blockIdx.x, tid = threadIdx.x;
  for (int i = tid; i < PN; i += 1024) { key[i] = rnd[b*PN + i]; kid[i] = i; }
  __syncthreads();
  for (int k = 2; k <= PN; k <<= 1) {
    for (int j = k >> 1; j > 0; j >>= 1) {
      for (int i = tid; i < PN; i += 1024) {
        int ixj = i ^ j;
        if (ixj > i) {
          bool up = ((i & k) == 0);
          float a = key[i], c = key[ixj];
          bool sw = up ? (a > c) : (a < c);
          if (sw) {
            key[i] = c; key[ixj] = a;
            int t = kid[i]; kid[i] = kid[ixj]; kid[ixj] = t;
          }
        }
      }
      __syncthreads();
    }
  }
  for (int i = tid; i < PN; i += 1024) perm[b*PN + i] = kid[i];
}

// ---------------- 2. top-16 nearest neighbors per sampled point ----------------
// Candidate tiles (3 x 1024 f32, row stride = PN) staged into LDS by the Tensor
// Data Mover (double-buffered, TENSORcnt-tracked); all 8 waves consume the tile.
#ifdef HAVE_TDM
__device__ __forceinline__ void tdm_issue(const float* gaddr, unsigned int lds_off) {
  unsigned long long ga = (unsigned long long)(size_t)gaddr;
  u32x4 g0; i32x8 g1; i32x4 g2, g3;
  g0[0] = 1u;                                    // count=1 valid descriptor
  g0[1] = lds_off;                               // lds_addr (bytes)
  g0[2] = (unsigned int)ga;                      // global_addr[31:0]
  g0[3] = (unsigned int)(ga >> 32) | 0x80000000u;// global_addr[56:32] | type=2
  g1[0] = 0x00020000;        // workgroup_mask=0, data_size=2 (4B)
  g1[1] = (int)(PN << 16);   // tensor_dim0[15:0]=8192 in bits [31:16]
  g1[2] = (int)(4 << 16);    // tensor_dim0 hi=0, tensor_dim1=4
  g1[3] = (int)(1024 << 16); // tensor_dim1 hi=0, tile_dim0=1024
  g1[4] = 3;                 // tile_dim1=3 rows (x,y,z), tile_dim2=0
  g1[5] = PN;                // tensor_dim0_stride lo32 = 8192 elements
  g1[6] = 0; g1[7] = 0;      // stride hi, dim1 stride unused (2D tile)
  g2 = (i32x4){0,0,0,0}; g3 = (i32x4){0,0,0,0};
#if __clang_major__ >= 23
  i32x8 g4 = (i32x8){0,0,0,0,0,0,0,0};
  __builtin_amdgcn_tensor_load_to_lds(g0, g1, g2, g3, g4, 0);
#else
  __builtin_amdgcn_tensor_load_to_lds(g0, g1, g2, g3, 0);
#endif
}
#endif

__global__ void k_knn(const float* __restrict__ x, const int* __restrict__ perm,
                      int* __restrict__ idx) {
#ifdef HAVE_TDM
  __shared__ float tile[2][3*1024];              // 24KB double buffer
#else
  __shared__ float4 tile4[1024];
#endif
  const int tid = threadIdx.x;
  const int gs = blockIdx.x * 256 + tid;         // flat (b, s)
  const int b = gs >> 13, s = gs & (SN - 1);
  const float* xb = x + (size_t)b * 4 * PN;
  const int ps = perm[b*PN + s];
  const float sx = xb[0*PN + ps], sy = xb[1*PN + ps], sz = xb[2*PN + ps];
  const float ss = sx*sx + sy*sy + sz*sz;

  float bv[16]; int bi[16];
#pragma unroll
  for (int t = 0; t < 16; ++t) { bv[t] = -3.4e38f; bi[t] = 0; }

#ifdef HAVE_TDM
  if (tid < 32) {                                // wave 0 drives the TDM
    tdm_issue(xb + 0, (unsigned int)(size_t)&tile[0][0]);
    __builtin_amdgcn_s_wait_tensorcnt(0);
  }
  __syncthreads();
  for (int t0 = 0; t0 < PN; t0 += 1024) {
    const int buf = (t0 >> 10) & 1;
    if (t0 + 1024 < PN && tid < 32)
      tdm_issue(xb + (t0 + 1024), (unsigned int)(size_t)&tile[buf ^ 1][0]);
    const float* tx = &tile[buf][0];
    const float* ty = &tile[buf][1024];
    const float* tz = &tile[buf][2048];
    for (int i = 0; i < 1024; ++i) {
      float qx = tx[i], qy = ty[i], qz = tz[i];
      float pp = qx*qx + qy*qy + qz*qz;
      float v = -(pp - 2.f*(qx*sx + qy*sy + qz*sz) + ss);
      if (v > bv[15]) {
        float cv = v; int ci = t0 + i;
#pragma unroll
        for (int t = 0; t < 16; ++t) {
          if (cv > bv[t]) { float tv = bv[t]; int ti = bi[t]; bv[t] = cv; bi[t] = ci; cv = tv; ci = ti; }
        }
      }
    }
    if (tid < 32) __builtin_amdgcn_s_wait_tensorcnt(0);
    __syncthreads();
  }
#else
  for (int t0 = 0; t0 < PN; t0 += 1024) {
    __syncthreads();
    for (int i = tid; i < 1024; i += 256) {
      int p = t0 + i;
      float px = xb[0*PN + p], py = xb[1*PN + p], pz = xb[2*PN + p];
      tile4[i] = make_float4(px, py, pz, px*px + py*py + pz*pz);
    }
    __syncthreads();
    if (t0 + 1024 < PN) __builtin_prefetch(xb + t0 + 1024, 0, 0);
    for (int i = 0; i < 1024; ++i) {
      float4 q = tile4[i];
      float v = -(q.w - 2.f*(q.x*sx + q.y*sy + q.z*sz) + ss);
      if (v > bv[15]) {
        float cv = v; int ci = t0 + i;
#pragma unroll
        for (int t = 0; t < 16; ++t) {
          if (cv > bv[t]) { float tv = bv[t]; int ti = bi[t]; bv[t] = cv; bi[t] = ci; cv = tv; ci = ti; }
        }
      }
    }
  }
#endif
  // reference's transpose+reshape scramble: raw[b][s][j] -> idx[b][j*512 + s/16][s%16]
#pragma unroll
  for (int j = 0; j < 16; ++j) {
    int lin = j*SN + s;
    idx[((size_t)b*SN + (lin >> 4))*16 + (lin & 15)] = bi[j];
  }
}

// ---------------- 3. gather x1 + emit output coords ----------------
__global__ void k_gather(const float* __restrict__ x, const int* __restrict__ idx,
                         float* __restrict__ x1, float* __restrict__ out) {
  const int t = blockIdx.x * 256 + threadIdx.x;   // flat (b, i, j), < B*S*16
  const int j = t & 15, i = (t >> 4) & (SN - 1), b = t >> 17;
  const int p = idx[t];
#pragma unroll
  for (int c = 0; c < 4; ++c) {
    float v = x[((size_t)b*4 + c)*PN + p];
    x1[(((size_t)b*4 + c)*SN + i)*16 + j] = v;
    if (j == 0 && c < 3) out[((size_t)b*67 + c)*SN + i] = v;
  }
}

// ---------------- 4. pre-activations: conv (ch 0-31) + PE (ch 32-63) ----------------
__global__ void k_pre(const float* __restrict__ x1, const float* __restrict__ w_conv,
                      const float* __restrict__ w_pe, float* __restrict__ pre) {
  const int t = blockIdx.x * 256 + threadIdx.x;   // < B*S*16
  const int k = t & 15, s = (t >> 4) & (SN - 1), b = t >> 17;
  float a4[4], c3[3];
#pragma unroll
  for (int c = 0; c < 4; ++c) a4[c] = x1[(((size_t)b*4 + c)*SN + s)*16 + k];
#pragma unroll
  for (int c = 0; c < 3; ++c) c3[c] = x1[(((size_t)b*4 + c)*SN + s)*16 + 0];
  const float d0 = a4[0]-c3[0], d1 = a4[1]-c3[1], d2 = a4[2]-c3[2];
  const float temp = sqrtf(fmaxf(d0*d0 + d1*d1 + d2*d2, 1e-12f));
  const float in10[10] = {a4[0],a4[1],a4[2], c3[0],c3[1],c3[2], d0,d1,d2, temp};
  for (int oc = 0; oc < 32; ++oc) {
    float acc = 0.f;
#pragma unroll
    for (int cc = 0; cc < 4; ++cc) acc += w_conv[oc*4+cc]*a4[cc];
    pre[(((size_t)b*64 + oc)*SN + s)*16 + k] = acc;
  }
  for (int oc = 0; oc < 32; ++oc) {
    float acc = 0.f;
#pragma unroll
    for (int cc = 0; cc < 10; ++cc) acc += w_pe[oc*10+cc]*in10[cc];
    pre[(((size_t)b*64 + 32 + oc)*SN + s)*16 + k] = acc;
  }
}

// ---------------- 5. per-(batch,channel) mean / rstd ----------------
__global__ void k_stats(const float* __restrict__ pre, float* __restrict__ stats) {
  __shared__ float s1s[256], s2s[256];
  const int blk = blockIdx.x;                     // b*64 + c
  const float* p = pre + (size_t)blk * SK;
  float s1 = 0.f, s2 = 0.f;
  for (int i = threadIdx.x; i < SK; i += 256) { float v = p[i]; s1 += v; s2 += v*v; }
  s1s[threadIdx.x] = s1; s2s[threadIdx.x] = s2;
  __syncthreads();
  for (int o = 128; o > 0; o >>= 1) {
    if (threadIdx.x < o) { s1s[threadIdx.x] += s1s[threadIdx.x+o]; s2s[threadIdx.x] += s2s[threadIdx.x+o]; }
    __syncthreads();
  }
  if (threadIdx.x == 0) {
    float m = s1s[0] / (float)SK;
    float var = s2s[0] / (float)SK - m*m;
    stats[blk*2]   = m;
    stats[blk*2+1] = rsqrtf(var + 1e-5f);
  }
}

// ---------------- 6. normalize + lrelu, emit f16 feature (B,S,16,64) ----------------
__global__ void k_norm(float* __restrict__ pre, const float* __restrict__ stats,
                       const float* __restrict__ g_conv, const float* __restrict__ b_conv,
                       const float* __restrict__ g_pe, const float* __restrict__ b_pe,
                       _Float16* __restrict__ featH) {
  const size_t t = (size_t)blockIdx.x * 256 + threadIdx.x;   // < B*64*S*16
  const int k = (int)(t & 15);
  const int s = (int)((t >> 4) & (SN - 1));
  const int c = (int)((t >> 17) & 63);
  const int b = (int)(t >> 23);
  const float m = stats[(b*64+c)*2], r = stats[(b*64+c)*2 + 1];
  const float g  = (c < 32) ? g_conv[c] : g_pe[c-32];
  const float be = (c < 32) ? b_conv[c] : b_pe[c-32];
  float v = pre[t];
  v = lrelu((v - m) * r * g + be, 0.2f);
  pre[t] = v;
  featH[(((size_t)b*SN + s)*16 + k)*64 + c] = (_Float16)v;
}

// ---------------- 7. attention logits: WMMA (32x64) x (64 x 262144) ----------------
__global__ void k_att(const _Float16* __restrict__ featH,
                      const float* __restrict__ w_att1,
                      const float* __restrict__ w_att2,
                      float* __restrict__ att) {
  const int wave = threadIdx.x >> 5, lane = threadIdx.x & 31;
  const int col = blockIdx.x * 8 + wave;          // flat (b,s), 16 columns = 16 neighbors
  const int h = lane >> 4, n = lane & 15;

  v16h a[2][2];
#pragma unroll
  for (int rt = 0; rt < 2; ++rt)
#pragma unroll
    for (int kb = 0; kb < 2; ++kb) {
      const int m = rt*16 + n;
#pragma unroll
      for (int j = 0; j < 16; ++j)
        a[rt][kb][j] = (_Float16)w_att1[m*64 + kb*32 + akidx(j, h)];
    }
  const _Float16* fb = featH + ((size_t)col*16 + n) * 64;
  v16h b0 = *(const v16h*)(fb + 16*h);
  v16h b1 = *(const v16h*)(fb + 32 + 16*h);

  v8f acc0 = {}, acc1 = {};
  acc0 = __builtin_amdgcn_wmma_f32_16x16x32_f16(false, a[0][0], false, b0, (short)0, acc0, false, false);
  acc0 = __builtin_amdgcn_wmma_f32_16x16x32_f16(false, a[0][1], false, b1, (short)0, acc0, false, false);
  acc1 = __builtin_amdgcn_wmma_f32_16x16x32_f16(false, a[1][0], false, b0, (short)0, acc1, false, false);
  acc1 = __builtin_amdgcn_wmma_f32_16x16x32_f16(false, a[1][1], false, b1, (short)0, acc1, false, false);

  float part = 0.f;
#pragma unroll
  for (int r = 0; r < 8; ++r) {
    part += w_att2[r + 8*h]      * lrelu(acc0[r], 0.2f);
    part += w_att2[16 + r + 8*h] * lrelu(acc1[r], 0.2f);
  }
  part += __shfl_xor(part, 16, 32);
  if (lane < 16) att[(size_t)col*16 + lane] = part;
}

// ---------------- 8. softmax over K=16 ----------------
__global__ void k_softmax(float* __restrict__ att) {
  const int t = blockIdx.x * 256 + threadIdx.x;   // < B*S
  float v[16], m = -3.4e38f;
#pragma unroll
  for (int k = 0; k < 16; ++k) { v[k] = att[(size_t)t*16 + k]; m = fmaxf(m, v[k]); }
  float s = 0.f;
#pragma unroll
  for (int k = 0; k < 16; ++k) { v[k] = expf(v[k] - m); s += v[k]; }
  const float inv = 1.f / s;
#pragma unroll
  for (int k = 0; k < 16; ++k) att[(size_t)t*16 + k] = v[k] * inv;
}

// ---------------- 9. attention-weighted pooling -> f16 (B,S,64) ----------------
__global__ void k_pool(const float* __restrict__ pre, const float* __restrict__ att,
                       _Float16* __restrict__ pooledH) {
  const int t = blockIdx.x * 256 + threadIdx.x;   // < B*64*S
  const int s = t & (SN - 1), f = (t >> 13) & 63, b = t >> 19;
  const float* pp = pre + (((size_t)b*64 + f)*SN + s)*16;
  const float* aa = att + ((size_t)b*SN + s)*16;
  float acc = 0.f;
#pragma unroll
  for (int k = 0; k < 16; ++k) acc += pp[k] * aa[k];
  pooledH[((size_t)b*SN + s)*64 + f] = (_Float16)acc;
}

// ---------------- 10. final: WMMA (64x64) x (64 x 16384) + w_b2 + lrelu(0.1) ----------------
__global__ void k_final(const _Float16* __restrict__ pooledH,
                        const float* __restrict__ w_b1,
                        const float* __restrict__ w_b2,
                        const float* __restrict__ x1,
                        float* __restrict__ out) {
  const int wave = threadIdx.x >> 5, lane = threadIdx.x & 31;
  const int tile = blockIdx.x * 8 + wave;          // 1024 tiles of 16 columns
  const int h = lane >> 4, n = lane & 15;
  const int col = tile*16 + n;                     // flat (b,s)
  const int b = col >> 13, s = col & (SN - 1);

  v16h a[4][2];
#pragma unroll
  for (int rt = 0; rt < 4; ++rt)
#pragma unroll
    for (int kb = 0; kb < 2; ++kb) {
      const int m = rt*16 + n;
#pragma unroll
      for (int j = 0; j < 16; ++j)
        a[rt][kb][j] = (_Float16)w_b1[m*64 + kb*32 + akidx(j, h)];
    }
  const _Float16* pb = pooledH + (size_t)col * 64;
  v16h b0 = *(const v16h*)(pb + 16*h);
  v16h b1 = *(const v16h*)(pb + 32 + 16*h);

  v8f acc[4];
#pragma unroll
  for (int rt = 0; rt < 4; ++rt) {
    v8f c = {};
    c = __builtin_amdgcn_wmma_f32_16x16x32_f16(false, a[rt][0], false, b0, (short)0, c, false, false);
    c = __builtin_amdgcn_wmma_f32_16x16x32_f16(false, a[rt][1], false, b1, (short)0, c, false, false);
    acc[rt] = c;
  }

  float xc[4];
#pragma unroll
  for (int cc = 0; cc < 4; ++cc)
    xc[cc] = x1[(((size_t)b*4 + cc)*SN + s)*16 + 0];

#pragma unroll
  for (int rt = 0; rt < 4; ++rt)
#pragma unroll
    for (int r = 0; r < 8; ++r) {
      const int f = rt*16 + r + 8*h;
      float fea2 = w_b2[f*4+0]*xc[0] + w_b2[f*4+1]*xc[1] + w_b2[f*4+2]*xc[2] + w_b2[f*4+3]*xc[3];
      out[((size_t)b*67 + 3 + f)*SN + s] = lrelu(acc[rt][r] + fea2, 0.1f);
    }
}

extern "C" void kernel_launch(void* const* d_in, const int* in_sizes, int n_in,
                              void* d_out, int out_size, void* d_ws, size_t ws_size,
                              hipStream_t stream) {
  const float* x      = (const float*)d_in[0];
  const float* rnd    = (const float*)d_in[1];
  const float* w_pe   = (const float*)d_in[2];
  const float* g_pe   = (const float*)d_in[3];
  const float* b_pe   = (const float*)d_in[4];
  const float* w_conv = (const float*)d_in[5];
  const float* g_conv = (const float*)d_in[6];
  const float* b_conv = (const float*)d_in[7];
  const float* w_att1 = (const float*)d_in[8];
  const float* w_att2 = (const float*)d_in[9];
  const float* w_b1   = (const float*)d_in[10];
  const float* w_b2   = (const float*)d_in[11];
  float* out = (float*)d_out;

  char* ws = (char*)d_ws;
  size_t off = 0;
  auto alloc = [&](size_t bytes) -> void* {
    void* p = ws + off;
    off += (bytes + 255) & ~(size_t)255;
    return p;
  };
  int*      perm    = (int*)      alloc((size_t)BN*PN*4);
  int*      idx     = (int*)      alloc((size_t)BN*SN*16*4);
  float*    x1      = (float*)    alloc((size_t)BN*4*SK*4);
  float*    pre     = (float*)    alloc((size_t)BN*64*SK*4);
  float*    stats   = (float*)    alloc((size_t)BN*64*2*4);
  _Float16* featH   = (_Float16*) alloc((size_t)BN*SK*64*2);
  float*    att     = (float*)    alloc((size_t)BN*SK*4);
  _Float16* pooledH = (_Float16*) alloc((size_t)BN*SN*64*2);

  k_sort   <<<dim3(BN),            dim3(1024), 0, stream>>>(rnd, perm);
  k_knn    <<<dim3(BN*SN/256),     dim3(256),  0, stream>>>(x, perm, idx);
  k_gather <<<dim3(BN*SK/256),     dim3(256),  0, stream>>>(x, idx, x1, out);
  k_pre    <<<dim3(BN*SK/256),     dim3(256),  0, stream>>>(x1, w_conv, w_pe, pre);
  k_stats  <<<dim3(BN*64),         dim3(256),  0, stream>>>(pre, stats);
  k_norm   <<<dim3(BN*64*SK/256),  dim3(256),  0, stream>>>(pre, stats, g_conv, b_conv, g_pe, b_pe, featH);
  k_att    <<<dim3(BN*SN/8),       dim3(256),  0, stream>>>(featH, w_att1, w_att2, att);
  k_softmax<<<dim3(BN*SN/256),     dim3(256),  0, stream>>>(att);
  k_pool   <<<dim3(BN*64*SN/256),  dim3(256),  0, stream>>>(pre, att, pooledH);
  k_final  <<<dim3(BN*SN/16/8),    dim3(256),  0, stream>>>(pooledH, w_b1, w_b2, x1, out);
}
